// CustomDeepseekDBOModel_31894427140773
// MI455X (gfx1250) — matrix operand order repaired and checked by reading, not compiled, not run on hardware
//
#include <hip/hip_runtime.h>
#include <hip/hip_bf16.h>

// ---- problem constants (match reference) ----
#define T_TOK   1024
#define H_DIM   2048
#define E_NUM   16
#define I_DIM   704
#define IS_DIM  1408
#define ACT_LD  (E_NUM * I_DIM + IS_DIM)   // 12672 fused activation width
#define TOPK    6
#define NGROUP  4

#define KC      64       // K-slab staged through LDS per iteration
#define KPAD    72       // LDS K pitch (144 B: 16B-aligned, conflict-free stride)

typedef __attribute__((ext_vector_type(16))) __bf16 v16bf;
typedef __attribute__((ext_vector_type(8)))  __bf16 v8bf;
typedef __attribute__((ext_vector_type(2)))  __bf16 v2bf;
typedef __attribute__((ext_vector_type(8)))  float  v8f;

// ---------------------------------------------------------------------------
// WMMA fragment loaders per CDNA5 ISA 7.12.2 (wave32).
// A (16x32 bf16, MxK): lane = half*16+row; element j -> K = 16*(j>>3)+half*8+(j&7)
// => two contiguous 8-elem (16 B) runs per lane.
// ---------------------------------------------------------------------------
__device__ __forceinline__ v16bf load_a_frag(const __bf16* tile, int ld, int k0) {
  const int lane = threadIdx.x & 31;
  const int half = lane >> 4;
  const int row  = lane & 15;
  const __bf16* p = tile + (size_t)row * ld + k0 + half * 8;
  v8bf lo = *(const v8bf*)p;
  v8bf hi = *(const v8bf*)(p + 16);
  v16bf r;
#pragma unroll
  for (int j = 0; j < 8; ++j) { r[j] = lo[j]; r[8 + j] = hi[j]; }
  return r;
}

// B (32x16 bf16, KxN) from LDS staged col-major: Bs[col*KPAD + k], k contiguous.
// Lane holds column colbase+(lane&15), K = s*32 + 16*(j>>3) + half*8 + (j&7)  -> 2x ds_load_b128.
__device__ __forceinline__ v16bf load_b_lds(const __bf16* Bs, int colbase, int s) {
  const int lane = threadIdx.x & 31;
  const int half = lane >> 4;
  const int col  = colbase + (lane & 15);
  const __bf16* p = Bs + col * KPAD + s * 32 + half * 8;
  v8bf lo = *(const v8bf*)p;
  v8bf hi = *(const v8bf*)(p + 16);
  v16bf r;
#pragma unroll
  for (int j = 0; j < 8; ++j) { r[j] = lo[j]; r[8 + j] = hi[j]; }
  return r;
}

// ---------------------------------------------------------------------------
// K2: f32 -> bf16 conversion of activations (vectorized b128 loads)
// ---------------------------------------------------------------------------
__global__ __launch_bounds__(256) void cvt_bf16_kernel(const float* __restrict__ x,
                                                       __bf16* __restrict__ xb) {
  size_t i = ((size_t)blockIdx.x * blockDim.x + threadIdx.x) * 8;
  float4 a = *(const float4*)(x + i);
  float4 b = *(const float4*)(x + i + 4);
  v8bf o;
  o[0] = (__bf16)a.x; o[1] = (__bf16)a.y; o[2] = (__bf16)a.z; o[3] = (__bf16)a.w;
  o[4] = (__bf16)b.x; o[5] = (__bf16)b.y; o[6] = (__bf16)b.z; o[7] = (__bf16)b.w;
  *(v8bf*)(xb + i) = o;
}

// ---------------------------------------------------------------------------
// K1: router: logits -> softmax -> grouped top-k -> combine[T,E] (x2.5 folded)
// ---------------------------------------------------------------------------
__global__ __launch_bounds__(256) void router_kernel(const float* __restrict__ x,
                                                     const float* __restrict__ wg,
                                                     float* __restrict__ combine) {
  const int t = blockIdx.x;
  __shared__ float sc[E_NUM];
  if (threadIdx.x < E_NUM) sc[threadIdx.x] = 0.f;
  __syncthreads();

  float acc[E_NUM];
#pragma unroll
  for (int e = 0; e < E_NUM; ++e) acc[e] = 0.f;
  for (int h = threadIdx.x; h < H_DIM; h += blockDim.x) {
    float xv = x[(size_t)t * H_DIM + h];
    const float* wrow = wg + (size_t)h * E_NUM;
#pragma unroll
    for (int e = 0; e < E_NUM; ++e) acc[e] += xv * wrow[e];
  }
#pragma unroll
  for (int e = 0; e < E_NUM; ++e) atomicAdd(&sc[e], acc[e]);  // ds_add_f32
  __syncthreads();

  if (threadIdx.x == 0) {
    float p[E_NUM];
    float mx = sc[0];
    for (int e = 1; e < E_NUM; ++e) mx = fmaxf(mx, sc[e]);
    float sum = 0.f;
    for (int e = 0; e < E_NUM; ++e) { p[e] = __expf(sc[e] - mx); sum += p[e]; }
    float inv = __builtin_amdgcn_rcpf(sum);
    for (int e = 0; e < E_NUM; ++e) p[e] *= inv;
    float gs[NGROUP];
    for (int g = 0; g < NGROUP; ++g) {
      float m = p[g * 4];
      for (int j = 1; j < 4; ++j) m = fmaxf(m, p[g * 4 + j]);
      gs[g] = m;
    }
    int g0 = 0; for (int g = 1; g < NGROUP; ++g) if (gs[g] > gs[g0]) g0 = g;
    gs[g0] = -1.f;
    int g1 = 0; for (int g = 1; g < NGROUP; ++g) if (gs[g] > gs[g1]) g1 = g;
    float m[E_NUM];
    for (int e = 0; e < E_NUM; ++e) {
      int g = e >> 2;
      m[e] = (g == g0 || g == g1) ? p[e] : -1.f;
    }
    float w[E_NUM]; for (int e = 0; e < E_NUM; ++e) w[e] = 0.f;
    float tsum = 0.f;
    for (int k = 0; k < TOPK; ++k) {
      int j = 0;
      for (int e = 1; e < E_NUM; ++e) if (m[e] > m[j]) j = e;
      if (m[j] > 0.f) { w[j] = m[j]; tsum += m[j]; }
      m[j] = -1.f;
    }
    float s = (tsum > 0.f) ? (2.5f / tsum) : 0.f;  // fold ROUTED_SCALING here
    for (int e = 0; e < E_NUM; ++e) combine[(size_t)t * E_NUM + e] = w[e] * s;
  }
}

// ---------------------------------------------------------------------------
// K3/K4: act[:, base + e*iw + n] = silu(x@Wg) * (x@Wu) * combine
// Block = 256 thr = 8 waves; WG tile = 64 tokens x (32 g-cols + 32 u-cols).
// Waves: 4 m-subtiles x 2 n-subtiles -> each staged B byte feeds 4 waves.
// Per KC=64 slab: coalesced f32 b64 loads -> cvt_pk -> LDS col-major bf16;
// each wave does 4 WMMAs (2 k-steps x {g,u}) from ds_load_b128 pairs.
// ---------------------------------------------------------------------------
__global__ __launch_bounds__(256) void gateup_silu_kernel(
    const __bf16* __restrict__ xb, const float* __restrict__ W,
    const float* __restrict__ combine,  // nullptr => shared expert (scale 1)
    __bf16* __restrict__ act,
    int ntiles_per_e, long wstride_e, int two_i, int iw, int colbase0) {
  __shared__ __bf16 Bs[64 * KPAD];      // 9 KB
  const int e    = blockIdx.y / ntiles_per_e;
  const int nt   = blockIdx.y % ntiles_per_e;
  const int m0   = blockIdx.x * 64;
  const int tid  = threadIdx.x;
  const int wave = tid >> 5;
  const int mi   = wave & 3;            // 0..3
  const int ni   = wave >> 2;           // 0..1
  const int gbase = nt * 32;

  const float*  We    = W + (size_t)e * wstride_e;
  const __bf16* atile = xb + (size_t)(m0 + mi * 16) * H_DIM;

  v8f accg = {};
  v8f accu = {};
  for (int k0 = 0; k0 < H_DIM; k0 += KC) {
    __syncthreads();                    // previous slab fully consumed
    // ---- stage W[k0..k0+63][g:gbase..+31 | u:iw+gbase..+31] -> Bs ----
#pragma unroll
    for (int i = 0; i < 4; ++i) {
      int q  = tid + 256 * i;           // 1024 quads (2 cols x 2 rows each)
      int c2 = (q & 31) * 2;            // local col 0..62 (even)
      int k2 = (q >> 5) * 2;            // local k   0..62 (even)
      int gc = (c2 < 32) ? (gbase + c2) : (iw + gbase + (c2 - 32));
      const float* p0 = We + (size_t)(k0 + k2) * two_i + gc;
      float2 r0 = *(const float2*)p0;
      float2 r1 = *(const float2*)(p0 + two_i);
      v2bf w0; w0[0] = (__bf16)r0.x; w0[1] = (__bf16)r1.x;  // v_cvt_pk_bf16_f32
      v2bf w1; w1[0] = (__bf16)r0.y; w1[1] = (__bf16)r1.y;
      *(v2bf*)&Bs[c2 * KPAD + k2]       = w0;
      *(v2bf*)&Bs[(c2 + 1) * KPAD + k2] = w1;
    }
    {   // branchless prefetch of next slab (wraps to k=0 at the end)
      int kpf = (k0 + KC) & (H_DIM - 1);
      __builtin_prefetch(We + (size_t)kpf * two_i + gbase + (tid & 31), 0, 3);
    }
    __syncthreads();
    // ---- compute: 2 k-steps x (g,u) per wave ----
#pragma unroll
    for (int s = 0; s < 2; ++s) {
      v16bf a  = load_a_frag(atile, H_DIM, k0 + s * 32);
      v16bf bg = load_b_lds(Bs, ni * 16, s);
      v16bf bu = load_b_lds(Bs, 32 + ni * 16, s);
      accg = __builtin_amdgcn_wmma_f32_16x16x32_bf16(false, a, false, bg,
                                                     (short)0, accg, false, false);
      accu = __builtin_amdgcn_wmma_f32_16x16x32_bf16(false, a, false, bu,
                                                     (short)0, accu, false, false);
    }
  }

  const int lane = threadIdx.x & 31;
  const int half = lane >> 4;
  const int col  = lane & 15;
  const size_t cb = (size_t)colbase0 + (size_t)e * iw + gbase + ni * 16 + col;
#pragma unroll
  for (int r = 0; r < 8; ++r) {
    const int row = m0 + mi * 16 + half * 8 + r;     // C/D layout: M = half*8+r
    float g = accg[r], u = accu[r];
    float av = g * __builtin_amdgcn_rcpf(1.f + __expf(-g)) * u;   // silu(g)*u
    float s  = combine ? combine[(size_t)row * E_NUM + e] : 1.f;
    act[(size_t)row * ACT_LD + cb] = (__bf16)(av * s);
  }
}

// ---------------------------------------------------------------------------
// K5: out[T,H] = act[T,12672] @ [w_down(stacked E*I rows) ; w_shared_down].
// WG tile 64 x 128; waves: 4 m-subtiles x 2 n-groups of 4 n-tiles.
// Per KC=64 slab: stage 64x128 f32 weights -> LDS bf16 (each byte feeds 4
// waves); each wave: 8 WMMAs per slab with A-fragment reuse across 4 n-tiles.
// ---------------------------------------------------------------------------
__global__ __launch_bounds__(256) void down_combine_kernel(
    const __bf16* __restrict__ act, const float* __restrict__ wdn,
    const float* __restrict__ wsdn, float* __restrict__ out) {
  __shared__ __bf16 Bs[128 * KPAD];     // 18 KB
  const int m0   = blockIdx.x * 64;
  const int n0   = blockIdx.y * 128;
  const int tid  = threadIdx.x;
  const int wave = tid >> 5;
  const int mi   = wave & 3;            // 0..3
  const int ng   = (wave >> 2) * 64;    // 0 or 64

  const __bf16* atile = act + (size_t)(m0 + mi * 16) * ACT_LD;
  v8f acc[4] = {};

  for (int k0 = 0; k0 < ACT_LD; k0 += KC) {
    const float* Bg;
    int kk;
    if (k0 < E_NUM * I_DIM) { Bg = wdn + n0;  kk = k0; }
    else                    { Bg = wsdn + n0; kk = k0 - E_NUM * I_DIM; }
    __syncthreads();
    // ---- stage Bg[kk..kk+63][0..127] -> Bs (transposed, bf16) ----
#pragma unroll
    for (int i = 0; i < 8; ++i) {
      int q  = tid + 256 * i;           // 2048 quads
      int c2 = (q & 63) * 2;            // local col 0..126 (even)
      int k2 = (q >> 6) * 2;            // local k   0..62  (even)
      const float* p0 = Bg + (size_t)(kk + k2) * H_DIM + c2;
      float2 r0 = *(const float2*)p0;
      float2 r1 = *(const float2*)(p0 + H_DIM);
      v2bf w0; w0[0] = (__bf16)r0.x; w0[1] = (__bf16)r1.x;
      v2bf w1; w1[0] = (__bf16)r0.y; w1[1] = (__bf16)r1.y;
      *(v2bf*)&Bs[c2 * KPAD + k2]       = w0;
      *(v2bf*)&Bs[(c2 + 1) * KPAD + k2] = w1;
    }
    __syncthreads();
    // ---- compute: 2 k-steps x 4 n-tiles ----
#pragma unroll
    for (int s = 0; s < 2; ++s) {
      v16bf a = load_a_frag(atile, ACT_LD, k0 + s * 32);
#pragma unroll
      for (int j = 0; j < 4; ++j) {
        v16bf b = load_b_lds(Bs, ng + j * 16, s);
        acc[j] = __builtin_amdgcn_wmma_f32_16x16x32_bf16(false, a, false, b,
                                                         (short)0, acc[j], false, false);
      }
    }
  }

  const int lane = threadIdx.x & 31;
  const int half = lane >> 4;
  const int col  = lane & 15;
#pragma unroll
  for (int j = 0; j < 4; ++j) {
#pragma unroll
    for (int r = 0; r < 8; ++r) {
      const int row = m0 + mi * 16 + half * 8 + r;
      out[(size_t)row * H_DIM + n0 + ng + j * 16 + col] = acc[j][r];
    }
  }
}

// ---------------------------------------------------------------------------
extern "C" void kernel_launch(void* const* d_in, const int* in_sizes, int n_in,
                              void* d_out, int out_size, void* d_ws, size_t ws_size,
                              hipStream_t stream) {
  const float* x    = (const float*)d_in[0];  // [T,H]
  const float* wg   = (const float*)d_in[1];  // [H,E]
  const float* wgu  = (const float*)d_in[2];  // [E,H,2I]
  const float* wdn  = (const float*)d_in[3];  // [E,I,H] == stacked rows [E*I, H]
  const float* wsgu = (const float*)d_in[4];  // [H,2*IS]
  const float* wsdn = (const float*)d_in[5];  // [IS,H]
  float* out = (float*)d_out;                 // [T,H] f32

  // workspace layout (~30 MB):
  char* wsb = (char*)d_ws;
  __bf16* xb      = (__bf16*)wsb;                                        // T*H*2   = 4 MB
  float*  combine = (float*)(wsb + (size_t)T_TOK * H_DIM * 2);           // T*E*4   = 64 KB
  __bf16* actbuf  = (__bf16*)(wsb + (size_t)T_TOK * H_DIM * 2
                                  + (size_t)T_TOK * E_NUM * 4);          // T*12672*2 = 26 MB

  dim3 blk(256);

  // 1) x -> bf16
  cvt_bf16_kernel<<<dim3((T_TOK * H_DIM) / (256 * 8)), blk, 0, stream>>>(x, xb);

  // 2) router -> combine weights (2.5x folded in)
  router_kernel<<<dim3(T_TOK), blk, 0, stream>>>(x, wg, combine);

  // 3) routed experts gate_up + SiLU + combine: grid.y = E * (704/32) = 352
  gateup_silu_kernel<<<dim3(T_TOK / 64, E_NUM * (I_DIM / 32)), blk, 0, stream>>>(
      xb, wgu, combine, actbuf,
      /*ntiles_per_e=*/I_DIM / 32, /*wstride_e=*/(long)H_DIM * 2 * I_DIM,
      /*two_i=*/2 * I_DIM, /*iw=*/I_DIM, /*colbase0=*/0);

  // 4) shared expert gate_up + SiLU: grid.y = 1408/32 = 44
  gateup_silu_kernel<<<dim3(T_TOK / 64, IS_DIM / 32), blk, 0, stream>>>(
      xb, wsgu, nullptr, actbuf,
      /*ntiles_per_e=*/IS_DIM / 32, /*wstride_e=*/0,
      /*two_i=*/2 * IS_DIM, /*iw=*/IS_DIM, /*colbase0=*/E_NUM * I_DIM);

  // 5) fused down-projection over routed + shared: out written exactly once
  down_combine_kernel<<<dim3(T_TOK / 64, H_DIM / 128), blk, 0, stream>>>(
      actbuf, wdn, wsdn, out);
}
